// LGnet_mem_16234976379374
// MI455X (gfx1250) — compile-verified, matrix operand order checked by reading, not run on hardware
//
#include <hip/hip_runtime.h>
#include <hip/hip_bf16.h>

// ---- problem dims ----
#define Bsz 256
#define Tt  200
#define Dd  128
#define Hh  256
#define Oo  128
#define Mm  64
#define KG  320   // M + H      (gate GEMM K)
#define KL  384   // 2D + O     (ls GEMM K)

// LDS row strides (bf16 elems); row stride * 2 bytes is a multiple of 16 (b128-aligned)
#define ALS_S (KL + 8)   // 392 -> 784 B/row
#define CMB_S (KG + 8)   // 328 -> 656 B/row

#define Z8 {0.f, 0.f, 0.f, 0.f, 0.f, 0.f, 0.f, 0.f}

typedef __attribute__((ext_vector_type(16))) __bf16        bf16x16;
typedef __attribute__((ext_vector_type(8)))  float         floatx8;
typedef __attribute__((ext_vector_type(4)))  unsigned int  uint4v;

union FragU {
    uint4v  q[2];
    bf16x16 v;
};

__device__ __forceinline__ unsigned short f2bf(float f) {
    unsigned int u = __float_as_uint(f);
    u += 0x7FFFu + ((u >> 16) & 1u);       // round-to-nearest-even
    return (unsigned short)(u >> 16);
}

__device__ __forceinline__ float sigmoidf(float x) {
    return 1.0f / (1.0f + __expf(-x));
}

// One 16x32 bf16 A/B fragment (ISA 16-bit layout): lane holds row; VGPR0-3 =
// K[k0+half*8 .. +7], VGPR4-7 = K[k0+16+half*8 .. +7]; two 16B contiguous loads.
__device__ __forceinline__ bf16x16 load_frag(const unsigned short* base, int stride,
                                             int row, int k0, int half) {
    const unsigned short* p = base + row * stride + k0 + half * 8;
    FragU f;
    f.q[0] = *(const uint4v*)(p);
    f.q[1] = *(const uint4v*)(p + 16);
    return f.v;
}

__device__ __forceinline__ floatx8 wmma_bf16(bf16x16 a, bf16x16 b, floatx8 c) {
    return __builtin_amdgcn_wmma_f32_16x16x32_bf16(false, a, false, b, (short)0, c,
                                                   false, false);
}

// Elementwise GRU-D style imputation for one (row r, feature d) at timestep t.
__device__ __forceinline__ void e1_elem(const float* __restrict__ inp,
                                        const float* __restrict__ Xmean,
                                        long chs, int b, int t, int d,
                                        float wzd, float bzd, float wzpd, float bzpd,
                                        unsigned short* Als, int r) {
    long base = ((long)b * 6) * chs + (long)t * Dd + d;
    float x   = inp[base];
    float xl  = inp[base + chs];
    float mk  = inp[base + 2 * chs];
    float dl  = inp[base + 3 * chs];
    float xlb = inp[base + 4 * chs];
    float dlb = inp[base + 5 * chs];
    float xm  = Xmean[((long)b * Tt + t) * Dd + d];

    float az  = dl  * wzd  + bzd;   az  = az  > 0.f ? az  : 0.f;
    float azp = dlb * wzpd + bzpd;  azp = azp > 0.f ? azp : 0.f;
    float dz  = __expf(-az);
    float dzp = __expf(-azp);

    float z  = mk * x + (1.f - mk) * (dz  * xl  + (1.f - dz ) * xm);
    float zp = mk * x + (1.f - mk) * (dzp * xlb + (1.f - dzp) * xm);
    Als[r * ALS_S + d]       = f2bf(z);
    Als[r * ALS_S + 128 + d] = f2bf(zp);
}

// ---- weight f32 -> bf16 pre-pack ----
__global__ void cvt_bf16_kernel(const float* __restrict__ src,
                                unsigned short* __restrict__ dst, int n) {
    int i = blockIdx.x * blockDim.x + threadIdx.x;
    if (i < n) dst[i] = f2bf(src[i]);
}

// ---- persistent recurrence: one workgroup = 16 batch rows, full T loop ----
__launch_bounds__(512, 1)
__global__ void lgnet_kernel(const float* __restrict__ inp,    // [B,6,T,D]
                             const float* __restrict__ Xmean,  // [B,T,D]
                             const float* __restrict__ Wz,
                             const float* __restrict__ bz,
                             const float* __restrict__ Wzp,
                             const float* __restrict__ bzp,
                             const float* __restrict__ bq,
                             const float* __restrict__ bi,
                             const float* __restrict__ bfg,
                             const float* __restrict__ bo,
                             const float* __restrict__ bc,
                             const float* __restrict__ bfc,
                             const unsigned short* __restrict__ Wq_b,   // [64,384]  bf16
                             const unsigned short* __restrict__ Wi_b,   // [256,320] bf16
                             const unsigned short* __restrict__ Wf_b,
                             const unsigned short* __restrict__ Wo_b,
                             const unsigned short* __restrict__ Wc_b,
                             const unsigned short* __restrict__ Wfc_b,  // [128,256] bf16
                             float* __restrict__ out)                    // [B,T,O]
{
    extern __shared__ char smem[];
    unsigned short* Als  = (unsigned short*)smem;          // [16][ALS_S]  z|zp|x_i
    unsigned short* Comb = Als + 16 * ALS_S;               // [16][CMB_S]  ls|h
    unsigned short* WqL  = Comb + 16 * CMB_S;              // [64][384]  LDS weight cache
    unsigned short* WfcL = WqL + Mm * KL;                  // [128][256] LDS weight cache

    const int tid  = threadIdx.x;
    const int wave = tid >> 5;
    const int lane = tid & 31;
    const int half = lane >> 4;
    const int l15  = lane & 15;
    const int b0   = blockIdx.x * 16;
    const long chs = (long)Tt * Dd;

    // --- loop-invariant per-thread scalars ---
    const int   d_e  = tid & 127;
    const float wzd  = Wz [d_e * Dd + d_e];
    const float bzd  = bz [d_e];
    const float wzpd = Wzp[d_e * Dd + d_e];
    const float bzpd = bzp[d_e];

    const int   ng   = wave * 16 + l15;            // this wave's gate column
    const float biv  = bi [ng];
    const float bfv  = bfg[ng];
    const float bov  = bo [ng];
    const float bcv  = bc [ng];
    const float bqv  = bq [(wave & 3) * 16 + l15]; // used by waves 0-3
    const float bfcv = bfc[(wave & 7) * 16 + l15]; // used by waves 0-7

    // --- stage Wq and Wfc into LDS (b128 copies) ---
    {
        const uint4v* s1 = (const uint4v*)Wq_b;   uint4v* t1 = (uint4v*)WqL;
        for (int i = tid; i < Mm * KL / 8; i += 512) t1[i] = s1[i];
        const uint4v* s2 = (const uint4v*)Wfc_b;  uint4v* t2 = (uint4v*)WfcL;
        for (int i = tid; i < Oo * Hh / 8; i += 512) t2[i] = s2[i];
    }
    // --- t=0 state: h = 0 (bf16), x_i = bfc, c = 0 (registers) ---
    for (int idx = tid; idx < 16 * 256; idx += 512) {
        int r = idx >> 8, col = idx & 255;
        Comb[r * CMB_S + Mm + col] = 0;
    }
    for (int idx = tid; idx < 16 * 128; idx += 512) {
        int r = idx >> 7, col = idx & 127;
        Als[r * ALS_S + 256 + col] = f2bf(bfc[col]);
    }
    floatx8 creg = Z8;

    // --- prologue E1 for t = 0 (all 512 threads, 4 elems each) ---
    {
        int r0 = tid >> 7;
        #pragma unroll
        for (int j = 0; j < 4; ++j)
            e1_elem(inp, Xmean, chs, b0 + r0 + 4 * j, 0, d_e,
                    wzd, bzd, wzpd, bzpd, Als, r0 + 4 * j);
    }
    __syncthreads();

    for (int t = 0; t < Tt; ++t) {
        floatx8 ai = Z8, af = Z8, ao = Z8, ac = Z8;

        // ==== P1: waves 0-3 additionally compute ls -> Comb[:,0:64];
        //          all 16 waves accumulate the h-part of their gate tile ====
        if (wave < 4) {
            const int nq = (wave & 3) * 16 + l15;
            floatx8 acc0 = Z8, acc1 = Z8;
            // double-buffered fragment slots: ping-pong on fixed registers
            bf16x16 aF[2], wF[2];
            aF[0] = load_frag(Als, ALS_S, l15, 0, half);
            wF[0] = load_frag(WqL, KL, nq, 0, half);
            #pragma unroll
            for (int kc = 0; kc < KL / 32; ++kc) {
                const int cur = kc & 1;
                if (kc + 1 < KL / 32) {
                    aF[cur ^ 1] = load_frag(Als, ALS_S, l15, (kc + 1) * 32, half);
                    wF[cur ^ 1] = load_frag(WqL, KL, nq, (kc + 1) * 32, half);
                }
                if (cur) acc1 = wmma_bf16(aF[1], wF[1], acc1);
                else     acc0 = wmma_bf16(aF[0], wF[0], acc0);
            }
            #pragma unroll
            for (int r = 0; r < 8; ++r)
                Comb[(r + half * 8) * CMB_S + nq] = f2bf(acc0[r] + acc1[r] + bqv);
        }
        // gate h-part: K = 64..319 (h from previous step).
        // Partial unroll keeps at most 16 global b128 loads in flight and bounds
        // VGPR pressure (avoids allocator copy-storms from over-hoisting).
        #pragma unroll 2
        for (int kc = 2; kc < 10; ++kc) {
            bf16x16 a = load_frag(Comb, CMB_S, l15, kc * 32, half);
            ai = wmma_bf16(a, load_frag(Wi_b, KG, ng, kc * 32, half), ai);
            af = wmma_bf16(a, load_frag(Wf_b, KG, ng, kc * 32, half), af);
            ao = wmma_bf16(a, load_frag(Wo_b, KG, ng, kc * 32, half), ao);
            ac = wmma_bf16(a, load_frag(Wc_b, KG, ng, kc * 32, half), ac);
        }
        __syncthreads();

        // ==== P2: ls-part of gates + in-register LSTM cell update -> h ====
        {
            bf16x16 a0 = load_frag(Comb, CMB_S, l15, 0,  half);
            bf16x16 a1 = load_frag(Comb, CMB_S, l15, 32, half);
            ai = wmma_bf16(a0, load_frag(Wi_b, KG, ng, 0, half), ai);
            af = wmma_bf16(a0, load_frag(Wf_b, KG, ng, 0, half), af);
            ao = wmma_bf16(a0, load_frag(Wo_b, KG, ng, 0, half), ao);
            ac = wmma_bf16(a0, load_frag(Wc_b, KG, ng, 0, half), ac);
            ai = wmma_bf16(a1, load_frag(Wi_b, KG, ng, 32, half), ai);
            af = wmma_bf16(a1, load_frag(Wf_b, KG, ng, 32, half), af);
            ao = wmma_bf16(a1, load_frag(Wo_b, KG, ng, 32, half), ao);
            ac = wmma_bf16(a1, load_frag(Wc_b, KG, ng, 32, half), ac);
        }
        #pragma unroll
        for (int r = 0; r < 8; ++r) {              // same lane holds i,f,o,ct at (m,n)
            float gi = sigmoidf(ai[r] + biv);
            float gf = sigmoidf(af[r] + bfv);
            float go = sigmoidf(ao[r] + bov);
            float gc = tanhf   (ac[r] + bcv);
            float c = gf * creg[r] + gi * gc;
            creg[r] = c;
            Comb[(r + half * 8) * CMB_S + Mm + ng] = f2bf(go * tanhf(c));
        }
        __syncthreads();

        // ==== P3: waves 0-7: out GEMM (writes out + next x_i);
        //          waves 8-15: elementwise z/zp for step t+1 (disjoint Als cols) ====
        if (wave < 8) {
            const int n = (wave & 7) * 16 + l15;
            floatx8 acc0 = Z8, acc1 = Z8;
            bf16x16 aF[2], wF[2];
            aF[0] = load_frag(Comb, CMB_S, l15, Mm, half);
            wF[0] = load_frag(WfcL, Hh, n, 0, half);
            #pragma unroll
            for (int kc = 0; kc < Hh / 32; ++kc) {
                const int cur = kc & 1;
                if (kc + 1 < Hh / 32) {
                    aF[cur ^ 1] = load_frag(Comb, CMB_S, l15, Mm + (kc + 1) * 32, half);
                    wF[cur ^ 1] = load_frag(WfcL, Hh, n, (kc + 1) * 32, half);
                }
                if (cur) acc1 = wmma_bf16(aF[1], wF[1], acc1);
                else     acc0 = wmma_bf16(aF[0], wF[0], acc0);
            }
            #pragma unroll
            for (int r = 0; r < 8; ++r) {
                int m = r + half * 8;
                float v = acc0[r] + acc1[r] + bfcv;
                out[((long)(b0 + m) * Tt + t) * Oo + n] = v;
                Als[m * ALS_S + 256 + n] = f2bf(v);      // x_i for step t+1
            }
        } else if (t + 1 < Tt) {
            int u  = tid - 256;
            int r0 = u >> 7;
            #pragma unroll
            for (int j = 0; j < 8; ++j) {
                int r = r0 + 2 * j;
                e1_elem(inp, Xmean, chs, b0 + r, t + 1, d_e,
                        wzd, bzd, wzpd, bzpd, Als, r);
            }
        }
        __syncthreads();
    }
}

extern "C" void kernel_launch(void* const* d_in, const int* in_sizes, int n_in,
                              void* d_out, int out_size, void* d_ws, size_t ws_size,
                              hipStream_t stream) {
    (void)in_sizes; (void)n_in; (void)out_size; (void)ws_size;

    unsigned short* ws    = (unsigned short*)d_ws;
    unsigned short* Wq_b  = ws;                    // 64*384
    unsigned short* Wi_b  = Wq_b  + 64 * KL;       // 256*320
    unsigned short* Wf_b  = Wi_b  + Hh * KG;
    unsigned short* Wo_b  = Wf_b  + Hh * KG;
    unsigned short* Wc_b  = Wo_b  + Hh * KG;
    unsigned short* Wfc_b = Wc_b  + Hh * KG;       // 128*256

    auto cvt = [&](int idx, unsigned short* dst, int n) {
        cvt_bf16_kernel<<<(n + 255) / 256, 256, 0, stream>>>((const float*)d_in[idx], dst, n);
    };
    cvt(6,  Wq_b,  64 * KL);
    cvt(8,  Wi_b,  Hh * KG);
    cvt(10, Wf_b,  Hh * KG);
    cvt(12, Wo_b,  Hh * KG);
    cvt(14, Wc_b,  Hh * KG);
    cvt(16, Wfc_b, Oo * Hh);

    size_t shbytes = (size_t)(16 * ALS_S + 16 * CMB_S + Mm * KL + Oo * Hh)
                   * sizeof(unsigned short);   // ~138 KB

    lgnet_kernel<<<Bsz / 16, 512, shbytes, stream>>>(
        (const float*)d_in[0],  (const float*)d_in[1],
        (const float*)d_in[2],  (const float*)d_in[3],
        (const float*)d_in[4],  (const float*)d_in[5],
        (const float*)d_in[7],  (const float*)d_in[9],
        (const float*)d_in[11], (const float*)d_in[13],
        (const float*)d_in[15], (const float*)d_in[17],
        Wq_b, Wi_b, Wf_b, Wo_b, Wc_b, Wfc_b,
        (float*)d_out);
}
